// GeometricTransformerLayer_43052752175157
// MI455X (gfx1250) — matrix operand order repaired and checked by reading, not compiled, not run on hardware
//
#include <hip/hip_runtime.h>
#include <hip/hip_bf16.h>

#define B_  2
#define S_  2048
#define D_  1024
#define H_  16
#define FF_ 4096
#define DH_ 64
#define T_  (B_*S_)

typedef __bf16 bf16_t;
typedef __attribute__((ext_vector_type(16))) __bf16 v16bf;
typedef __attribute__((ext_vector_type(8)))  __bf16 v8bf;
typedef __attribute__((ext_vector_type(8)))  float  v8f;
typedef __attribute__((ext_vector_type(4)))  unsigned int v4u;
typedef __attribute__((ext_vector_type(8)))  int v8i;
typedef __attribute__((ext_vector_type(4)))  int v4i;

static __device__ __forceinline__ v8f wmma_bf16(v16bf a, v16bf b, v8f c) {
  return __builtin_amdgcn_wmma_f32_16x16x32_bf16(false, a, false, b, (short)0, c, false, false);
}

// A fragment: 16x32 bf16 tile, row-major source. A lane's 16
// elements are k = 16*half + 0..15 (one contiguous run) -> two 16B vector loads.
static __device__ __forceinline__ v16bf load_afrag(const bf16_t* A, int lda, int row, int col, int lane) {
  int m = lane & 15, half = lane >> 4;
  const bf16_t* base = A + (size_t)(row + m) * lda + col + 16 * half;
  union { v16bf f; v8bf h[2]; } u;
  u.h[0] = *(const v8bf*)(base);
  u.h[1] = *(const v8bf*)(base + 8);
  return u.f;
}

// B fragment where B^T is stored row-major as [N][K] (ldbt = K-stride):
// lane's 16 K-values (K = e + 16*half) are contiguous -> one 32B vector load.
static __device__ __forceinline__ v16bf load_bfragT(const bf16_t* Bt, int ldbt, int krow, int col, int lane) {
  int nn = lane & 15, half = lane >> 4;
  return *(const v16bf*)(Bt + (size_t)(col + nn) * ldbt + krow + 16 * half);
}

// ---------------- Tensor Data Mover: 2D tile global -> LDS ----------------
// Builds a D# per CDNA5 ISA (group0 + group1, 2D, bf16).
// dims/strides in elements (data_size = 2 bytes). Tracked by TENSORcnt.
static __device__ __forceinline__ void tdm_load_2d(const void* gaddr, unsigned lds_off,
    unsigned tile_d0, unsigned tile_d1, unsigned tensor_d0, unsigned tensor_d1,
    unsigned long long stride0) {
  unsigned long long ga = (unsigned long long)gaddr;
  v4u g0;
  g0[0] = 1u;                                        // count=1, user descriptor
  g0[1] = lds_off;                                   // lds_addr (bytes)
  g0[2] = (unsigned)(ga & 0xFFFFFFFFu);              // global_addr[31:0]
  g0[3] = (unsigned)((ga >> 32) & 0x1FFFFFFu) | (2u << 30);  // addr[56:32] | type=2
  v8i g1;
  g1[0] = (int)(1u << 16);                           // data_size=1 -> 2 bytes
  g1[1] = (int)((tensor_d0 & 0xFFFFu) << 16);        // tensor_dim0[15:0] @ bit48
  g1[2] = (int)((tensor_d0 >> 16) | ((tensor_d1 & 0xFFFFu) << 16));
  g1[3] = (int)((tensor_d1 >> 16) | (tile_d0 << 16));
  g1[4] = (int)(tile_d1 & 0xFFFFu);                  // tile_dim1; tile_dim2=0
  g1[5] = (int)(stride0 & 0xFFFFFFFFull);            // tensor_dim0_stride[31:0]
  g1[6] = (int)((stride0 >> 32) & 0xFFFFull);        // stride[47:32]; dim1_stride=0
  g1[7] = 0;
  v4i z4 = {0, 0, 0, 0};
#if __clang_major__ >= 23
  v8i z8 = {0, 0, 0, 0, 0, 0, 0, 0};
  __builtin_amdgcn_tensor_load_to_lds(g0, g1, z4, z4, z8, 0);
#else
  __builtin_amdgcn_tensor_load_to_lds(g0, g1, z4, z4, 0);
#endif
}

// ---------------- weight transpose + f32->bf16 convert ----------------
__global__ __launch_bounds__(256) void cvtT_kernel(const float* __restrict__ in,
    bf16_t* __restrict__ out, int R, int C) {
  __shared__ float tile[32][33];
  int g  = blockIdx.z;
  int r0 = blockIdx.y * 32, c0 = blockIdx.x * 32;
  int tx = threadIdx.x & 31, ty = threadIdx.x >> 5;
  const float* src = in  + (size_t)g * R * C;
  bf16_t*      dst = out + (size_t)g * R * C;
#pragma unroll
  for (int rr = ty; rr < 32; rr += 8)
    tile[rr][tx] = src[(size_t)(r0 + rr) * C + c0 + tx];
  __syncthreads();
#pragma unroll
  for (int rr = ty; rr < 32; rr += 8)
    dst[(size_t)(c0 + rr) * R + r0 + tx] = (bf16_t)tile[tx][rr];
}

// ---------------- elementwise / reduction kernels ----------------

__global__ __launch_bounds__(256) void ln_bf16_kernel(const float* __restrict__ x,
    const float* __restrict__ g, const float* __restrict__ bta, bf16_t* __restrict__ out) {
  int t = blockIdx.x;
  const float* row = x + (size_t)t * D_;
  float s = 0.f, s2 = 0.f;
  for (int i = threadIdx.x; i < D_; i += 256) { float v = row[i]; s += v; s2 += v * v; }
  int wave = threadIdx.x >> 5, lane = threadIdx.x & 31;
#pragma unroll
  for (int m = 16; m >= 1; m >>= 1) { s += __shfl_xor(s, m); s2 += __shfl_xor(s2, m); }
  __shared__ float rs[8], rs2[8];
  if (lane == 0) { rs[wave] = s; rs2[wave] = s2; }
  __syncthreads();
  if (threadIdx.x < 32) {
    s  = (lane < 8) ? rs[lane]  : 0.f;
    s2 = (lane < 8) ? rs2[lane] : 0.f;
#pragma unroll
    for (int m = 4; m >= 1; m >>= 1) { s += __shfl_xor(s, m); s2 += __shfl_xor(s2, m); }
    if (lane == 0) { rs[0] = s; rs2[0] = s2; }
  }
  __syncthreads();
  float mean = rs[0] * (1.f / D_);
  float var  = rs2[0] * (1.f / D_) - mean * mean;
  float rstd = rsqrtf(var + 1e-5f);
  bf16_t* orow = out + (size_t)t * D_;
  for (int i = threadIdx.x; i < D_; i += 256)
    orow[i] = (bf16_t)((row[i] - mean) * rstd * g[i] + bta[i]);
}

__global__ void k2_kernel(const bf16_t* __restrict__ kb, float* __restrict__ k2, int rows) {
  int r = blockIdx.x * blockDim.x + threadIdx.x;
  if (r < rows) {
    const v8bf* p = (const v8bf*)(kb + (size_t)r * DH_);
    float s = 0.f;
#pragma unroll
    for (int j = 0; j < 8; ++j) {
      v8bf c = p[j];
#pragma unroll
      for (int e = 0; e < 8; ++e) { float v = (float)c[e]; s += v * v; }
    }
    k2[r] = s;
  }
}

__global__ void kpos_kernel(const float* __restrict__ curv, float* __restrict__ out) {
  int i = threadIdx.x;
  if (i < H_) {
    float c = curv[i];
    out[i] = (c > 20.f) ? c : log1pf(__expf(c));
  }
}

// ---------------- per-head QKV projection (WMMA) ----------------
__global__ __launch_bounds__(256) void qkv_kernel(const bf16_t* __restrict__ hnb,
    const bf16_t* __restrict__ Wqt, const bf16_t* __restrict__ Wkt, const bf16_t* __restrict__ Wvt,
    const float* __restrict__ bq, const float* __restrict__ bk, const float* __restrict__ bv,
    bf16_t* __restrict__ qb, bf16_t* __restrict__ kb, bf16_t* __restrict__ vtb) {
  int h = blockIdx.y;
  int wave = threadIdx.x >> 5, lane = threadIdx.x & 31;
  int trow0 = blockIdx.x * 128 + wave * 16;
  int nn = lane & 15, half = lane >> 4;
  const v8f vz = {0.f,0.f,0.f,0.f,0.f,0.f,0.f,0.f};

  v16bf a0 = load_afrag(hnb, D_, trow0, h * DH_,      lane);
  v16bf a1 = load_afrag(hnb, D_, trow0, h * DH_ + 32, lane);
  const bf16_t* Wqh = Wqt + (size_t)h * DH_ * DH_;
  const bf16_t* Wkh = Wkt + (size_t)h * DH_ * DH_;
  const bf16_t* Wvh = Wvt + (size_t)h * DH_ * DH_;

  int bb = trow0 >> 11;
  int s0 = (trow0 & (S_ - 1)) + 8 * half;
  int bh = bb * H_ + h;

#pragma unroll
  for (int ct = 0; ct < 4; ++ct) {
    v8f aq = vz, ak = vz, av = vz;
    aq = wmma_bf16(a0, load_bfragT(Wqh, DH_, 0,  ct * 16, lane), aq);
    aq = wmma_bf16(a1, load_bfragT(Wqh, DH_, 32, ct * 16, lane), aq);
    ak = wmma_bf16(a0, load_bfragT(Wkh, DH_, 0,  ct * 16, lane), ak);
    ak = wmma_bf16(a1, load_bfragT(Wkh, DH_, 32, ct * 16, lane), ak);
    av = wmma_bf16(a0, load_bfragT(Wvh, DH_, 0,  ct * 16, lane), av);
    av = wmma_bf16(a1, load_bfragT(Wvh, DH_, 32, ct * 16, lane), av);
    int e = ct * 16 + nn;
    float bqv = bq[h * DH_ + e], bkv = bk[h * DH_ + e], bvv = bv[h * DH_ + e];
    v8bf vs;
#pragma unroll
    for (int i = 0; i < 8; ++i) {
      int t = trow0 + i + 8 * half;
      int ss = t & (S_ - 1);
      size_t idx = ((size_t)bh * S_ + ss) * DH_ + e;
      qb[idx] = (bf16_t)(aq[i] + bqv);
      kb[idx] = (bf16_t)(ak[i] + bkv);
      vs[i]   = (bf16_t)(av[i] + bvv);
    }
    *(v8bf*)(vtb + ((size_t)bh * DH_ + e) * S_ + s0) = vs;
  }
}

// ---------------- flash geometric attention (WMMA + TDM) ----------------
// scores = 2*scale*(q.k) - scale*k2[t]   (q2 term cancels in softmax).
// K/V tiles are DMA'd into LDS by the Tensor Data Mover once per block
// (all 8 waves share them), tracked with TENSORcnt.
__global__ __launch_bounds__(256) void attn_kernel(const bf16_t* __restrict__ qb,
    const bf16_t* __restrict__ kb, const bf16_t* __restrict__ vtb,
    const float* __restrict__ k2, const float* __restrict__ curv, bf16_t* __restrict__ ob) {
  int bh = blockIdx.y;
  int h = bh & (H_ - 1);
  int b = bh >> 4;
  int wave = threadIdx.x >> 5, lane = threadIdx.x & 31;
  int q0 = blockIdx.x * 128 + wave * 16;
  int nn = lane & 15, half = lane >> 4;
  const v8f vz = {0.f,0.f,0.f,0.f,0.f,0.f,0.f,0.f};

  const bf16_t* qh  = qb  + (size_t)bh * S_ * DH_;
  const bf16_t* kh  = kb  + (size_t)bh * S_ * DH_;
  const bf16_t* vth = vtb + (size_t)bh * DH_ * S_;
  const float*  k2h = k2  + (size_t)bh * S_;

  float c = curv[h];
  float kpos = (c > 20.f) ? c : log1pf(__expf(c));
  float scale = kpos * 0.125f;   // kpos / sqrt(64)

  v16bf aq0 = load_afrag(qh, DH_, q0, 0,  lane);
  v16bf aq1 = load_afrag(qh, DH_, q0, 32, lane);

  float mi[8], li[8];
  v8f o0 = vz, o1 = vz, o2 = vz, o3 = vz;
#pragma unroll
  for (int i = 0; i < 8; ++i) { mi[i] = -1e30f; li[i] = 0.f; }

  __shared__ bf16_t ldsK[32 * 64];     // [t_rel][dh]   K chunk
  __shared__ bf16_t ldsV[64 * 32];     // [dh][t_rel]   V^T chunk
  __shared__ bf16_t psh[8][16 * 32];   // per-wave P staging (C-layout -> A-layout)
  bf16_t* pw = psh[wave];

  for (int t0 = 0; t0 < S_; t0 += 32) {
    if (wave == 0) {
      // K tile: 32 rows x 64 els, row stride 64 els
      tdm_load_2d(kh + (size_t)t0 * DH_, (unsigned)(size_t)(void*)ldsK,
                  64u, 32u, 64u, (unsigned)S_, 64ull);
      // V^T tile: 64 rows x 32 els, row stride S els
      tdm_load_2d(vth + t0, (unsigned)(size_t)(void*)ldsV,
                  32u, 64u, (unsigned)S_, 64u, (unsigned long long)S_);
      __builtin_amdgcn_s_wait_tensorcnt(0);
    }
    __syncthreads();

    v8f s0 = vz, s1 = vz;
    s0 = wmma_bf16(aq0, load_bfragT(ldsK, DH_, 0,  0,  lane), s0);
    s0 = wmma_bf16(aq1, load_bfragT(ldsK, DH_, 32, 0,  lane), s0);
    s1 = wmma_bf16(aq0, load_bfragT(ldsK, DH_, 0,  16, lane), s1);
    s1 = wmma_bf16(aq1, load_bfragT(ldsK, DH_, 32, 16, lane), s1);

    float k2a = k2h[t0 + nn], k2c = k2h[t0 + 16 + nn];
    float va[8], vb2[8], rm[8];
#pragma unroll
    for (int i = 0; i < 8; ++i) {
      va[i]  = 2.f * scale * s0[i] - scale * k2a;
      vb2[i] = 2.f * scale * s1[i] - scale * k2c;
      rm[i]  = fmaxf(va[i], vb2[i]);
    }
#pragma unroll
    for (int m = 8; m >= 1; m >>= 1) {
#pragma unroll
      for (int i = 0; i < 8; ++i) rm[i] = fmaxf(rm[i], __shfl_xor(rm[i], m));
    }
    float ps[8];
#pragma unroll
    for (int i = 0; i < 8; ++i) {
      float nm   = fmaxf(mi[i], rm[i]);
      float corr = __expf(mi[i] - nm);
      mi[i] = nm;
      float pa = __expf(va[i]  - nm);
      float pb = __expf(vb2[i] - nm);
      ps[i] = pa + pb;
      li[i] *= corr;
      o0[i] *= corr; o1[i] *= corr; o2[i] *= corr; o3[i] *= corr;
      int r = i + 8 * half;
      pw[r * 32 + nn]      = (bf16_t)pa;
      pw[r * 32 + 16 + nn] = (bf16_t)pb;
    }
#pragma unroll
    for (int m = 8; m >= 1; m >>= 1) {
#pragma unroll
      for (int i = 0; i < 8; ++i) ps[i] += __shfl_xor(ps[i], m);
    }
#pragma unroll
    for (int i = 0; i < 8; ++i) li[i] += ps[i];

    __syncthreads();
    v16bf ap = load_afrag(pw, 32, 0, 0, lane);
    o0 = wmma_bf16(ap, load_bfragT(ldsV, 32, 0, 0,  lane), o0);
    o1 = wmma_bf16(ap, load_bfragT(ldsV, 32, 0, 16, lane), o1);
    o2 = wmma_bf16(ap, load_bfragT(ldsV, 32, 0, 32, lane), o2);
    o3 = wmma_bf16(ap, load_bfragT(ldsV, 32, 0, 48, lane), o3);
    __syncthreads();   // all LDS reads done before next chunk's DMA
  }

  size_t obase = ((size_t)b * S_ + q0) * D_ + h * DH_;
#pragma unroll
  for (int i = 0; i < 8; ++i) {
    int r = i + 8 * half;
    float inv = 1.f / li[i];
    ob[obase + (size_t)r * D_ + nn]      = (bf16_t)(o0[i] * inv);
    ob[obase + (size_t)r * D_ + 16 + nn] = (bf16_t)(o1[i] * inv);
    ob[obase + (size_t)r * D_ + 32 + nn] = (bf16_t)(o2[i] * inv);
    ob[obase + (size_t)r * D_ + 48 + nn] = (bf16_t)(o3[i] * inv);
  }
}

// ---------------- generic bf16 WMMA GEMM with fused epilogue ----------------
__global__ __launch_bounds__(256) void gemm_kernel(const bf16_t* __restrict__ A,
    const bf16_t* __restrict__ Bt, const float* __restrict__ bias,
    const float* __restrict__ res, float* __restrict__ outf, bf16_t* __restrict__ outb,
    int M, int N, int K, int act) {
  int wave = threadIdx.x >> 5, lane = threadIdx.x & 31;
  int wr = wave & 1, wc = wave >> 1;
  int row0 = blockIdx.y * 128 + wr * 64;
  int col0 = blockIdx.x * 128 + wc * 32;
  int nn = lane & 15, half = lane >> 4;
  const v8f vz = {0.f,0.f,0.f,0.f,0.f,0.f,0.f,0.f};

  v8f acc[4][2];
#pragma unroll
  for (int r = 0; r < 4; ++r) { acc[r][0] = vz; acc[r][1] = vz; }

  for (int k0 = 0; k0 < K; k0 += 32) {
    if (k0 + 32 < K) {
      __builtin_prefetch(A  + (size_t)(row0 + nn) * K + k0 + 32, 0, 1);
      __builtin_prefetch(Bt + (size_t)(col0 + nn) * K + k0 + 32, 0, 1);
    }
    v16bf b0 = load_bfragT(Bt, K, k0, col0,      lane);
    v16bf b1 = load_bfragT(Bt, K, k0, col0 + 16, lane);
#pragma unroll
    for (int r = 0; r < 4; ++r) {
      v16bf af = load_afrag(A, K, row0 + 16 * r, k0, lane);
      acc[r][0] = wmma_bf16(af, b0, acc[r][0]);
      acc[r][1] = wmma_bf16(af, b1, acc[r][1]);
    }
  }

#pragma unroll
  for (int r = 0; r < 4; ++r) {
#pragma unroll
    for (int cc = 0; cc < 2; ++cc) {
      int col = col0 + cc * 16 + nn;
      float bv = bias ? bias[col] : 0.f;
#pragma unroll
      for (int i = 0; i < 8; ++i) {
        int row = row0 + 16 * r + i + 8 * half;
        float v = acc[r][cc][i] + bv;
        if (act == 1) v = 0.5f * v * (1.f + erff(v * 0.70710678118f));
        size_t idx = (size_t)row * N + col;
        if (res)  v += res[idx];
        if (outf) outf[idx] = v;
        if (outb) outb[idx] = (bf16_t)v;
      }
    }
  }
}

// ---------------- orchestration ----------------

extern "C" void kernel_launch(void* const* d_in, const int* in_sizes, int n_in,
                              void* d_out, int out_size, void* d_ws, size_t ws_size,
                              hipStream_t stream) {
  const float* x    = (const float*)d_in[0];
  const float* Wq   = (const float*)d_in[1];
  const float* bq   = (const float*)d_in[2];
  const float* Wk   = (const float*)d_in[3];
  const float* bk   = (const float*)d_in[4];
  const float* Wv   = (const float*)d_in[5];
  const float* bv   = (const float*)d_in[6];
  const float* curv = (const float*)d_in[7];
  const float* Wo   = (const float*)d_in[8];
  const float* bo   = (const float*)d_in[9];
  const float* ln1g = (const float*)d_in[10];
  const float* ln1b = (const float*)d_in[11];
  const float* ln2g = (const float*)d_in[12];
  const float* ln2b = (const float*)d_in[13];
  const float* W1   = (const float*)d_in[14];
  const float* b1   = (const float*)d_in[15];
  const float* W2   = (const float*)d_in[16];
  const float* b2   = (const float*)d_in[17];

  char* ws = (char*)d_ws;
  size_t off = 0;
  auto alloc = [&](size_t bytes) -> char* {
    char* p = ws + off;
    off += (bytes + 255) & ~(size_t)255;
    return p;
  };

  bf16_t* hnb  = (bf16_t*)alloc((size_t)T_ * D_ * 2);
  bf16_t* qb   = (bf16_t*)alloc((size_t)T_ * D_ * 2);
  bf16_t* kb   = (bf16_t*)alloc((size_t)T_ * D_ * 2);
  bf16_t* vtb  = (bf16_t*)alloc((size_t)T_ * D_ * 2);   // [bh][e][s] transposed
  float*  k2w  = (float*) alloc((size_t)B_ * H_ * S_ * 4);
  bf16_t* ob   = (bf16_t*)alloc((size_t)T_ * D_ * 2);
  float*  x1   = (float*) alloc((size_t)T_ * D_ * 4);
  bf16_t* hn2b = (bf16_t*)alloc((size_t)T_ * D_ * 2);
  bf16_t* ffb  = (bf16_t*)alloc((size_t)T_ * FF_ * 2);
  bf16_t* Wqt  = (bf16_t*)alloc((size_t)H_ * DH_ * DH_ * 2);
  bf16_t* Wkt  = (bf16_t*)alloc((size_t)H_ * DH_ * DH_ * 2);
  bf16_t* Wvt  = (bf16_t*)alloc((size_t)H_ * DH_ * DH_ * 2);
  bf16_t* Wot  = (bf16_t*)alloc((size_t)D_ * D_ * 2);
  bf16_t* W1t  = (bf16_t*)alloc((size_t)D_ * FF_ * 2);
  bf16_t* W2t  = (bf16_t*)alloc((size_t)FF_ * D_ * 2);
  (void)ws_size; (void)in_sizes; (void)n_in; (void)out_size;

  cvtT_kernel<<<dim3(2, 2, H_),    256, 0, stream>>>(Wq, Wqt, DH_, DH_);
  cvtT_kernel<<<dim3(2, 2, H_),    256, 0, stream>>>(Wk, Wkt, DH_, DH_);
  cvtT_kernel<<<dim3(2, 2, H_),    256, 0, stream>>>(Wv, Wvt, DH_, DH_);
  cvtT_kernel<<<dim3(32, 32, 1),   256, 0, stream>>>(Wo, Wot, D_, D_);
  cvtT_kernel<<<dim3(128, 32, 1),  256, 0, stream>>>(W1, W1t, D_, FF_);
  cvtT_kernel<<<dim3(32, 128, 1),  256, 0, stream>>>(W2, W2t, FF_, D_);

  ln_bf16_kernel<<<T_, 256, 0, stream>>>(x, ln1g, ln1b, hnb);

  qkv_kernel<<<dim3(T_ / 128, H_), 256, 0, stream>>>(hnb, Wqt, Wkt, Wvt, bq, bk, bv, qb, kb, vtb);

  k2_kernel<<<(B_ * H_ * S_) / 256, 256, 0, stream>>>(kb, k2w, B_ * H_ * S_);

  attn_kernel<<<dim3(S_ / 128, B_ * H_), 256, 0, stream>>>(qb, kb, vtb, k2w, curv, ob);

  gemm_kernel<<<dim3(D_ / 128, T_ / 128), 256, 0, stream>>>(
      ob, Wot, bo, x, x1, (bf16_t*)nullptr, T_, D_, D_, 0);

  ln_bf16_kernel<<<T_, 256, 0, stream>>>(x1, ln2g, ln2b, hn2b);

  gemm_kernel<<<dim3(FF_ / 128, T_ / 128), 256, 0, stream>>>(
      hn2b, W1t, b1, (const float*)nullptr, (float*)nullptr, ffb, T_, FF_, D_, 1);

  gemm_kernel<<<dim3(D_ / 128, T_ / 128), 256, 0, stream>>>(
      ffb, W2t, b2, x1, (float*)d_out, (bf16_t*)nullptr, T_, D_, FF_, 0);

  kpos_kernel<<<1, 32, 0, stream>>>(curv, (float*)d_out + (size_t)T_ * D_);
}
